// ContrastiveLoss_71768903516810
// MI455X (gfx1250) — matrix operand order repaired and checked by reading, compiled, not run on hardware
//
#include <hip/hip_runtime.h>
#include <hip/hip_bf16.h>

// ---------------------------------------------------------------------------
// ContrastiveLoss for MI455X (gfx1250, wave32, WMMA).
// Heavy path: single fused pass over video_feats upper-triangle (136 MB,
// HBM-bound ~6us @ 23.3TB/s): load f32 -> norm^2 on the fly -> bf16 LDS tile
// -> v_wmma_f32_16x16x32_bf16 GEMM against [qf rows | this-video tvf rows]
// -> post-scale scores by 1/||col|| -> masked exp-sum reductions -> f32
// global atomics. Small GEMMs are exact f32, parallelized one-block-per-row
// so no single-WGP serial tail dominates the HBM-bound main pass.
// ---------------------------------------------------------------------------

typedef __attribute__((ext_vector_type(16))) __bf16 v16bf;
typedef __attribute__((ext_vector_type(8)))  float  v8f;

#define BB   64      // batch (videos)
#define CC   256     // channels
#define DD   64      // 2D map side
#define TT   128     // sentences
#define PP   2080    // triu entries of 64x64
#define DSQ  4096    // D*D
#define INVT 10.0f   // 1/T_V == 1/T_Q == 1/0.1

// ---- workspace layout (bytes) ----------------------------------------------
#define OFF_LUT       0        // 2080 int : p -> r*64+c
#define OFF_SCATTER   8448     // 128 int
#define OFF_TOPK      8960     // 128 int
#define OFF_TSTART    9472     // 64 int
#define OFF_TCNT      9728     // 64 int
#define OFF_NEGQ      9984     // 64 f32   (inter_query neg exp-sums, atomic)
#define OFF_NEGINTRA  10240    // 128 f32  (intra_video neg exp-sums, atomic)
#define OFF_NEGIV     10752    // 128 f32
#define OFF_POSIV     11264    // 128 f32
#define OFF_LOSSIQ    11776    // 128 f32  (per-sentence intra_query loss)
#define OFF_QF        12288    // 64*256 f32 (normalized)
#define OFF_SF        77824    // 128*256 f32 (normalized)
#define OFF_TVF       208896   // 128*256 f32 (topk vectors, normalized)
#define OFF_QFB       339968   // 64*256 bf16
#define OFF_TVFB      372736   // 128*256 bf16
#define WS_NEEDED     438272

// ---------------------------------------------------------------------------
__global__ void init_kernel(const int* __restrict__ num_targets,
                            int* __restrict__ lut, int* __restrict__ scat,
                            int* __restrict__ tstart, int* __restrict__ tcnt,
                            float* __restrict__ negq, float* __restrict__ negintra)
{
    const int tid = threadIdx.x;
    if (tid < BB) {
        const int r = tid;
        const int base = r * DD - (r * (r - 1)) / 2;   // triu row start
        for (int c = r; c < DD; ++c) lut[base + (c - r)] = r * DD + c;
        negq[tid] = 0.f;
    }
    if (tid < TT) negintra[tid] = 0.f;
    if (tid == 0) {
        int t = 0;
        for (int b = 0; b < BB; ++b) {
            tstart[b] = t;
            int n = num_targets[b];
            tcnt[b] = n;                  // K == 1
            for (int i = 0; i < n && t < TT; ++i) scat[t++] = b;
        }
        while (t < TT) scat[t++] = BB - 1;
    }
}

// rows 0..63 -> qf, rows 64..191 -> sf ; one block per row, 256 threads
__global__ void normalize_kernel(const float* __restrict__ query,
                                 const float* __restrict__ sents,
                                 float* __restrict__ qf, float* __restrict__ sf,
                                 __bf16* __restrict__ qfb)
{
    __shared__ float red[CC];
    const int row = blockIdx.x;
    const int c   = threadIdx.x;
    const float v = (row < BB) ? query[row * CC + c] : sents[(row - BB) * CC + c];
    red[c] = v * v;
    __syncthreads();
    for (int s = CC / 2; s > 0; s >>= 1) {
        if (c < s) red[c] += red[c + s];
        __syncthreads();
    }
    const float inv = 1.f / fmaxf(sqrtf(red[0]), 1e-12f);
    const float o = v * inv;
    if (row < BB) { qf[row * CC + c] = o; qfb[row * CC + c] = (__bf16)o; }
    else         { sf[(row - BB) * CC + c] = o; }
}

// one block per sentence, 128 threads: argmax over triu of iou2ds[t]
__global__ void topk_kernel(const float* __restrict__ iou2ds,
                            const int* __restrict__ lut, int* __restrict__ topkp)
{
    __shared__ float bv_[128];
    __shared__ int   bp_[128];
    const int t = blockIdx.x, tid = threadIdx.x;
    float best = -1e30f; int bp = 0;
    for (int p = tid; p < PP; p += 128) {
        const float v = iou2ds[t * DSQ + lut[p]];
        if (v > best) { best = v; bp = p; }   // p ascending -> first occurrence
    }
    bv_[tid] = best; bp_[tid] = bp;
    __syncthreads();
    for (int s = 64; s > 0; s >>= 1) {
        if (tid < s) {
            const float v2 = bv_[tid + s]; const int p2 = bp_[tid + s];
            if (v2 > bv_[tid] || (v2 == bv_[tid] && p2 < bp_[tid])) { bv_[tid] = v2; bp_[tid] = p2; }
        }
        __syncthreads();
    }
    if (tid == 0) topkp[t] = bp_[0];
}

// one block per sentence, 256 threads: gather + normalize winning column
__global__ void build_tvf_kernel(const float* __restrict__ video,
                                 const int* __restrict__ lut,
                                 const int* __restrict__ scat,
                                 const int* __restrict__ topkp,
                                 float* __restrict__ tvf, __bf16* __restrict__ tvfb)
{
    __shared__ float red[CC];
    const int t = blockIdx.x, c = threadIdx.x;
    const int b = scat[t];
    const int off = lut[topkp[t]];
    const float v = video[((size_t)(b * CC + c)) * DSQ + off];
    red[c] = v * v;
    __syncthreads();
    for (int s = CC / 2; s > 0; s >>= 1) {
        if (c < s) red[c] += red[c + s];
        __syncthreads();
    }
    const float inv = 1.f / fmaxf(sqrtf(red[0]), 1e-12f);
    const float o = v * inv;
    tvf[t * CC + c] = o;
    tvfb[t * CC + c] = (__bf16)o;
}

// ---------------------------------------------------------------------------
// The fused WMMA pass: grid = (33 p-tiles of 64, 64 videos), 256 thr = 8 waves.
// A (80x256 bf16): rows 0..63 = qf, rows 64..64+cnt = this video's tvf rows.
// B (64x256 bf16): raw vf columns (norm applied post-GEMM via invn[col]).
// ---------------------------------------------------------------------------
#define LDA 258
#define LDB 258
#define PTILE 64

__global__ __launch_bounds__(256)
void big_gemm_kernel(const float* __restrict__ video,
                     const float* __restrict__ iou2d,
                     const int* __restrict__ lut,
                     const int* __restrict__ tstart_a, const int* __restrict__ tcnt_a,
                     const __bf16* __restrict__ qfb, const __bf16* __restrict__ tvfb,
                     float* __restrict__ negsum_q, float* __restrict__ intra_negsum)
{
    extern __shared__ char smem[];
    __bf16* Alds  = (__bf16*)smem;                  // 80 * LDA
    __bf16* Blds  = Alds + 80 * LDA;                // 64 * LDB
    float*  Slds  = (float*)(Blds + 64 * LDB);      // 80 * 64 scores
    float*  colsq = Slds + 80 * 64;                 // 64
    float*  invn  = colsq + 64;                     // 64
    float*  iouc  = invn + 64;                      // 64
    int*    vmask = (int*)(iouc + 64);              // 64

    const int tid = threadIdx.x;
    const int bv  = blockIdx.y;
    const int p0  = blockIdx.x * PTILE;
    const int tstart = tstart_a[bv];
    const int cnt    = min(tcnt_a[bv], 16);

    const int j  = tid & 63;        // column within tile
    const int ch = tid >> 6;        // 0..3 k-chunk of 64
    const int p  = p0 + j;
    const int valid = (p < PP) ? 1 : 0;
    const int off   = valid ? lut[p] : 0;

    if (tid < 64) {
        colsq[tid] = 0.f;
        vmask[tid] = valid;
        iouc[tid]  = valid ? iou2d[bv * DSQ + off] : 2.0f;
    }
    // A tile: qf rows then tvf rows (zero-padded to 80)
    for (int i = tid; i < BB * CC; i += 256)
        Alds[(i >> 8) * LDA + (i & 255)] = qfb[i];
    for (int i = tid; i < 16 * CC; i += 256) {
        const int j2 = i >> 8, c = i & 255;
        Alds[(BB + j2) * LDA + c] =
            (j2 < cnt) ? tvfb[(tstart + j2) * CC + c] : (__bf16)0.0f;
    }
    __syncthreads();

    // B tile: coalesced strided load of 64 columns, sum of squares on the fly
    {
        const float* vb = video + (size_t)bv * CC * DSQ + off;
        float ss = 0.f;
        const int cbeg = ch * 64;
        for (int c = cbeg; c < cbeg + 64; ++c) {
            const float v = valid ? vb[(size_t)c * DSQ] : 0.f;
            ss += v * v;
            Blds[j * LDB + c] = (__bf16)v;
        }
        atomicAdd(&colsq[j], ss);
    }
    __syncthreads();
    if (tid < 64) invn[tid] = 1.f / fmaxf(sqrtf(colsq[tid]), 1e-12f);
    __syncthreads();

    // --- WMMA: 5 M-tiles x 4 N-tiles, K=256 in 8 steps of 32 ---------------
    const int wave = tid >> 5;
    const int lane = tid & 31;
    const int lrow = lane & 15;
    const int ah = (lane >> 4) ? 8 : 0;      // A half-wave K offset
    const int bh = (lane >> 4) ? 8 : 0;      // B half-wave K offset
    for (int task = wave; task < 20; task += 8) {
        const int mt = task / 4;
        const int nt = task % 4;
        v8f acc = {};
        const __bf16* arow = &Alds[(mt * 16 + lrow) * LDA];
        const __bf16* bcol = &Blds[(nt * 16 + lrow) * LDB];
#pragma unroll
        for (int s = 0; s < 8; ++s) {
            const int base = s * 32;
            v16bf a, b;
#pragma unroll
            for (int i = 0; i < 8; ++i) {
                a[i]     = arow[base + ah + i];        // K = base+ah+0..7
                a[8 + i] = arow[base + 16 + ah + i];   // K = base+16+ah+0..7
            }
#pragma unroll
            for (int i = 0; i < 8; ++i) {
                b[i]     = bcol[base + bh + i];        // K = base+bh+0..7
                b[8 + i] = bcol[base + 16 + bh + i];   // K = base+16+bh+0..7
            }
            acc = __builtin_amdgcn_wmma_f32_16x16x32_bf16(
                false, a, false, b, (short)0, acc, false, false);
        }
#pragma unroll
        for (int jj = 0; jj < 8; ++jj) {
            const int m = mt * 16 + ((lane < 16) ? jj : jj + 8);
            Slds[m * 64 + nt * 16 + lrow] = acc[jj];
        }
    }
    __syncthreads();

    // --- masked exp-sum reductions, one row per thread ---------------------
    if (tid < 80) {
        const int m = tid;
        if (m < BB) {                      // inter_query negatives for query m
            float s = 0.f;
            for (int jj = 0; jj < 64; ++jj) {
                if (!vmask[jj]) continue;
                if (m == bv && iouc[jj] > 0.5f) continue;   // ~pos_mask
                s += expf(Slds[m * 64 + jj] * invn[jj] * INVT);
            }
            atomicAdd(&negsum_q[m], s);
        } else {                           // intra_video negatives for tvf row
            const int j2 = m - BB;
            if (j2 < cnt) {
                float s = 0.f;
                for (int jj = 0; jj < 64; ++jj) {
                    if (!vmask[jj]) continue;
                    if (!(iouc[jj] < 0.5f)) continue;
                    s += expf(Slds[m * 64 + jj] * invn[jj] * INVT);
                }
                atomicAdd(&intra_negsum[tstart + j2], s);
            }
        }
    }
}

// ---------------------------------------------------------------------------
// inter_video: one block per sentence t (128 blocks, 64 threads).
// Thread b computes tvf[t].qf[b] (tvf row staged in LDS); block-reduce the
// b != b_t exp-sum; thread b_t emits the positive score.
__global__ void inter_video_kernel(const float* __restrict__ tvf,
                                   const float* __restrict__ qf,
                                   const int* __restrict__ scat,
                                   float* __restrict__ pos_iv, float* __restrict__ neg_iv)
{
    __shared__ float tvs[CC];
    __shared__ float red[BB];
    const int t = blockIdx.x, tid = threadIdx.x;   // tid = candidate video b
    const int bt = scat[t];
    for (int c = tid; c < CC; c += BB) tvs[c] = tvf[t * CC + c];
    __syncthreads();
    const float* q = qf + tid * CC;
    float d = 0.f;
    for (int c = 0; c < CC; ++c) d += tvs[c] * q[c];
    if (tid == bt) pos_iv[t] = d;
    red[tid] = (tid == bt) ? 0.f : expf(d * INVT);
    __syncthreads();
    for (int s = BB / 2; s > 0; s >>= 1) {
        if (tid < s) red[tid] += red[tid + s];
        __syncthreads();
    }
    if (tid == 0) neg_iv[t] = red[0];
}

// intra_query: one block per sentence t (128 blocks, 128 threads).
// Thread j computes qf[b_t].sf[j] (qf row staged in LDS); masked exp-sum
// over scat[j] != b_t; per-sentence loss written to workspace.
__global__ void intra_query_kernel(const float* __restrict__ qf,
                                   const float* __restrict__ sf,
                                   const int* __restrict__ scat,
                                   float* __restrict__ loss_iq)
{
    __shared__ float qs[CC];
    __shared__ float red[TT];
    __shared__ float poss;
    const int t = blockIdx.x, tid = threadIdx.x;   // tid = other sentence j
    const int bt = scat[t];
    for (int c = tid; c < CC; c += TT) qs[c] = qf[bt * CC + c];
    __syncthreads();
    const float* s = sf + tid * CC;
    float d = 0.f;
    for (int c = 0; c < CC; ++c) d += qs[c] * s[c];
    if (tid == t) poss = d;                        // pos = sf[t].qf[b_t]
    red[tid] = (scat[tid] == bt) ? 0.f : expf(d * INVT);
    __syncthreads();
    for (int s2 = TT / 2; s2 > 0; s2 >>= 1) {
        if (tid < s2) red[tid] += red[tid + s2];
        __syncthreads();
    }
    if (tid == 0) {
        const float pv = poss * INVT;
        loss_iq[t] = -(pv - logf(expf(pv) + red[0]));
    }
}

// out[0..3]. 256 threads, single block (pure reduction, ~0.1 MFLOP).
__global__ void finalize_kernel(const float* __restrict__ pos_iv,
                                const float* __restrict__ neg_iv,
                                const float* __restrict__ negq,
                                const float* __restrict__ negintra,
                                const float* __restrict__ loss_iq,
                                const float* __restrict__ tvf,
                                const int* __restrict__ scat,
                                float* __restrict__ out)
{
    __shared__ float r0[256], r1[256], r2[256], r3[256];
    const int tid = threadIdx.x;
    float l0 = 0.f, l1 = 0.f, l3 = 0.f;
    if (tid < TT) {
        const float pv = pos_iv[tid] * INVT;
        l0 = -(pv - logf(expf(pv) + neg_iv[tid]));
        l1 = -(pv - logf(expf(pv) + negq[scat[tid]]));
        l3 = loss_iq[tid];
    }
    // intra_video: 64 videos x (ref r, partner pc) in {0,1}^2, nK = 2
    const int b = tid >> 2, rr = (tid >> 1) & 1, pc = tid & 1;
    const float* x = tvf + (2 * b + rr) * CC;
    const float* y = tvf + (2 * b + pc) * CC;
    float d = 0.f;
    for (int c = 0; c < CC; ++c) d += x[c] * y[c];
    const float dv = d * INVT;
    const float l2 = -(dv - logf(expf(dv) + negintra[2 * b + rr]));
    r0[tid] = l0; r1[tid] = l1; r2[tid] = l2; r3[tid] = l3;
    __syncthreads();
    for (int s = 128; s > 0; s >>= 1) {
        if (tid < s) {
            r0[tid] += r0[tid + s]; r1[tid] += r1[tid + s];
            r2[tid] += r2[tid + s]; r3[tid] += r3[tid + s];
        }
        __syncthreads();
    }
    if (tid == 0) {
        out[0] = r0[0] / (float)TT;
        out[1] = r1[0] / (float)TT;
        out[2] = r2[0] / 256.0f;
        out[3] = r3[0] / (float)TT;
    }
}

// ---------------------------------------------------------------------------
extern "C" void kernel_launch(void* const* d_in, const int* in_sizes, int n_in,
                              void* d_out, int out_size, void* d_ws, size_t ws_size,
                              hipStream_t stream)
{
    const float* video  = (const float*)d_in[0];
    const float* query  = (const float*)d_in[1];
    const float* sents  = (const float*)d_in[2];
    const float* iou2d  = (const float*)d_in[3];
    const float* iou2ds = (const float*)d_in[4];
    const int*   ntarg  = (const int*)d_in[5];
    float* out = (float*)d_out;

    char* ws = (char*)d_ws;
    int*    lut      = (int*)(ws + OFF_LUT);
    int*    scat     = (int*)(ws + OFF_SCATTER);
    int*    topkp    = (int*)(ws + OFF_TOPK);
    int*    tstart   = (int*)(ws + OFF_TSTART);
    int*    tcnt     = (int*)(ws + OFF_TCNT);
    float*  negq     = (float*)(ws + OFF_NEGQ);
    float*  negintra = (float*)(ws + OFF_NEGINTRA);
    float*  negiv    = (float*)(ws + OFF_NEGIV);
    float*  posiv    = (float*)(ws + OFF_POSIV);
    float*  lossiq   = (float*)(ws + OFF_LOSSIQ);
    float*  qf       = (float*)(ws + OFF_QF);
    float*  sf       = (float*)(ws + OFF_SF);
    float*  tvf      = (float*)(ws + OFF_TVF);
    __bf16* qfb      = (__bf16*)(ws + OFF_QFB);
    __bf16* tvfb     = (__bf16*)(ws + OFF_TVFB);

    init_kernel<<<1, 256, 0, stream>>>(ntarg, lut, scat, tstart, tcnt, negq, negintra);
    normalize_kernel<<<BB + TT, CC, 0, stream>>>(query, sents, qf, sf, qfb);
    topk_kernel<<<TT, 128, 0, stream>>>(iou2ds, lut, topkp);
    build_tvf_kernel<<<TT, CC, 0, stream>>>(video, lut, scat, topkp, tvf, tvfb);

    const int ptiles = (PP + PTILE - 1) / PTILE;                  // 33
    const size_t smem = (size_t)(80 * LDA + 64 * LDB) * sizeof(__bf16)
                      + (size_t)(80 * 64 + 64 * 3) * sizeof(float)
                      + (size_t)64 * sizeof(int);                 // ~95.8 KB
    big_gemm_kernel<<<dim3(ptiles, BB), 256, smem, stream>>>(
        video, iou2d, lut, tstart, tcnt, qfb, tvfb, negq, negintra);

    inter_video_kernel<<<TT, BB, 0, stream>>>(tvf, qf, scat, posiv, negiv);
    intra_query_kernel<<<TT, TT, 0, stream>>>(qf, sf, scat, lossiq);
    finalize_kernel<<<1, 256, 0, stream>>>(posiv, negiv, negq, negintra, lossiq,
                                           tvf, scat, out);
}
